// ScaledDotAttention_85066122264700
// MI455X (gfx1250) — matrix operand order repaired and checked by reading, not compiled, not run on hardware
//
#include <hip/hip_runtime.h>
#include <hip/hip_bf16.h>

// ---------------------------------------------------------------------------
// MI455X (gfx1250) fused projection + scaled-dot attention, bf16 WMMA path,
// with Tensor-Data-Mover (TDM) staging of GEMM B-tiles into LDS.
//
// Shapes: B=4, S=K=2048, H=1024.  ~120 GFLOP matmul, ~215 MB mandatory HBM
// traffic -> ~10us floor at 23.3 TB/s; bf16 V_WMMA_F32_16X16X32_BF16 keeps
// tensor math near that floor.  The GEMM kernels stage the shared 64x64 L-tile
// once per workgroup via tensor_load_to_lds (double-buffered, overlapped with
// WMMA, s_wait_tensorcnt + barrier sync), eliminating the 8x per-wave
// redundant global B reads of the naive version.
// ---------------------------------------------------------------------------

typedef __bf16 bf16_t;
typedef __attribute__((ext_vector_type(16))) __bf16 v16bf;
typedef __attribute__((ext_vector_type(8)))  __bf16 v8bf;
typedef __attribute__((ext_vector_type(8)))  float  v8f;
typedef unsigned int v4u __attribute__((ext_vector_type(4)));
typedef int          v8i __attribute__((ext_vector_type(8)));
typedef int          v4i __attribute__((ext_vector_type(4)));

union FragBF  { v16bf v; uint4 q[2]; };  // 16 bf16 = 8 VGPRs (WMMA A/B operand)
union PackBF8 { v8bf  v; uint4 q;    };  // 8 bf16  = one b128 store

static constexpr int Bn = 4, Sn = 2048, Kn = 2048, Hn = 1024;

// LDS row stride (bf16 elems) for TDM-staged 64-wide tiles: 64 data + 8 pad
// (pad_amount = 4 DWORDs every 32 DWORDs) -> 144B stride, conflict-staggered.
static constexpr int BROW = 72;

// ---------------------------------------------------------------------------
// TDM: load a 64(rows) x 64(cols) bf16 tile, row stride ldl elems, into LDS
// at byte offset lds_off with 16B padding per 128B row (ISA 8.3-8.6 D#).
// This toolchain exposes the 6-arg builtin:
//   (uint32x4 g0, int32x8 g1, int32x4 g2, int32x4 g3, int32x8 g4, i32 cpol)
// Groups 2..4 are zero: 2-D tile, no gather/iterate.
// ---------------------------------------------------------------------------
__device__ __forceinline__ void tdm_load_tile64(const bf16_t* gtile, int ldl,
                                                unsigned lds_off) {
  unsigned long long ga = (unsigned long long)gtile;
  v4u g0;
  g0.x = 1u;                                         // count=1 valid descriptor
  g0.y = lds_off;                                    // lds_addr [63:32]
  g0.z = (unsigned)(ga & 0xFFFFFFFFu);               // global_addr [95:64]
  g0.w = (unsigned)((ga >> 32) & 0x1FFFFFFu)         // global_addr [120:96]
       | 0x80000000u;                                // type=2 ("image")
  v8i g1;
  g1[0] = (1 << 16)                                  // data_size = 2 bytes
        | (1 << 20)                                  // pad_enable
        | (4 << 22)                                  // pad_interval: 32 DWORDs
        | (3 << 25);                                 // pad_amount: 4 DWORDs
  const unsigned td0 = 0x40000000u, td1 = 0x40000000u;  // huge dims: no OOB
  g1[1] = (int)((td0 & 0xFFFFu) << 16);              // tensor_dim0[15:0]
  g1[2] = (int)((td0 >> 16) | ((td1 & 0xFFFFu) << 16));
  g1[3] = (int)((td1 >> 16) | (64u << 16));          // tile_dim0 = 64
  g1[4] = 64;                                        // tile_dim1=64, tile_dim2=0
  unsigned long long st0 = (unsigned long long)ldl;  // dim0 stride (elems)
  g1[5] = (int)(st0 & 0xFFFFFFFFu);
  g1[6] = (int)((st0 >> 32) & 0xFFFFu);              // dim1 stride = 0 (2D)
  g1[7] = 0;
  v4i g2 = {0, 0, 0, 0};
  v4i g3 = {0, 0, 0, 0};
  v8i g4 = {0, 0, 0, 0, 0, 0, 0, 0};
  __builtin_amdgcn_tensor_load_to_lds(g0, g1, g2, g3, g4, 0);
}

// ---------------------------------------------------------------------------
// Elementwise f32 -> bf16 (8 elements / thread, b128 in/out).
// ---------------------------------------------------------------------------
__global__ void cast_f32_to_bf16(const float* __restrict__ src,
                                 bf16_t* __restrict__ dst, int n8) {
  int i = blockIdx.x * blockDim.x + threadIdx.x;
  if (i >= n8) return;
  const float4* s = (const float4*)src;
  float4 a = s[2 * i + 0];
  float4 b = s[2 * i + 1];
  PackBF8 o;
  o.v[0] = (bf16_t)a.x; o.v[1] = (bf16_t)a.y; o.v[2] = (bf16_t)a.z; o.v[3] = (bf16_t)a.w;
  o.v[4] = (bf16_t)b.x; o.v[5] = (bf16_t)b.y; o.v[6] = (bf16_t)b.z; o.v[7] = (bf16_t)b.w;
  ((uint4*)dst)[i] = o.q;
}

// ---------------------------------------------------------------------------
// Generic bf16 GEMM:  C(M,N) = A(M,Kd) * Bmat,  Bmat given as L(N,Kd) with
// L[n,kk] = Bmat[kk,n] (K-contiguous per output column = WMMA B-frag layout).
// B-side: TDM double-buffered 64x64 LDS tiles shared by all 8 waves.
// A-side: direct per-lane b128 global loads (no intra-WG redundancy).
// Each wave: 16 rows x 64 cols (4 C tiles); 8 waves/WG -> 128x64 per WG.
// ---------------------------------------------------------------------------
template <bool OUT_BF16, bool TRANS>
__global__ void gemm_bf16(const bf16_t* __restrict__ A,
                          const bf16_t* __restrict__ L,
                          void* __restrict__ Out,
                          int Kd, int lda, int ldl, int ldo, int batchRows,
                          unsigned long long aBatch,
                          unsigned long long lBatch,
                          unsigned long long oBatch) {
  __shared__ __align__(16) bf16_t bTile[2][64 * BROW];  // 2 x 9216 B
  __shared__ float tileLds[8][16 * 20];                 // store micro-transpose

  const int tid = threadIdx.x;
  const int l   = tid & 31;
  const int w   = tid >> 5;

  // WMMA bf16 lane mapping used here (wave32; A and B share one contiguous K order):
  const int am  = l & 15;          // A: row within tile
  const int ah8 = (l >> 4) * 8;    // A: contiguous K run [kk+2*ah8,+16)
  const int bn  = l & 15;          // B: output column
  const int bk  = (l >> 4) * 16;   // B: contiguous K run [kk+bk,+16)
  const int cn  = l & 15;          // C: column
  const int cm  = (l >> 4) * 8;    // C: rows cm..cm+7 in VGPR 0..7
  const int rr  = l >> 1;          // LDS readback: row
  const int rh  = (l & 1) * 8;     // LDS readback: 8-col half

  const int m0 = (blockIdx.y * 8 + w) * 16;
  const int n0 = blockIdx.x * 64;
  const size_t z = blockIdx.z;

  const bf16_t* arow  = A + z * aBatch + (size_t)(m0 + am) * lda;
  const bf16_t* lbase = L + z * lBatch + (size_t)n0 * ldl;  // tile row 0

  const unsigned ldsB0 = (unsigned)(unsigned long long)&bTile[0][0];
  const unsigned ldsB1 = (unsigned)(unsigned long long)&bTile[1][0];
  const int nch = Kd >> 6;  // 64-wide K chunks

  if (w == 0) {
    tdm_load_tile64(lbase, ldl, ldsB0);
    __builtin_amdgcn_s_wait_tensorcnt((short)0);
  }
  __syncthreads();

  v8f c[4] = {};
  for (int ch = 0; ch < nch; ++ch) {
    // Kick off next chunk's TDM before compute: overlaps DMA with WMMA.
    if (w == 0 && ch + 1 < nch)
      tdm_load_tile64(lbase + (size_t)(ch + 1) * 64, ldl,
                      ((ch + 1) & 1) ? ldsB1 : ldsB0);

    const bf16_t* bt = &bTile[ch & 1][0];
    const int kg = ch << 6;

    // A fragments for this chunk (two K=32 steps), batched so waits stagger.
    FragBF a0, a1;
    a0.q[0] = *(const uint4*)(arow + kg + 2 * ah8);
    a0.q[1] = *(const uint4*)(arow + kg + 8 + 2 * ah8);
    a1.q[0] = *(const uint4*)(arow + kg + 32 + 2 * ah8);
    a1.q[1] = *(const uint4*)(arow + kg + 40 + 2 * ah8);
    if (kg + 128 < Kd) __builtin_prefetch(arow + kg + 128, 0, 3);

    // B fragments from LDS (all loads issued before any WMMA).
    FragBF bb0[4], bb1[4];
#pragma unroll
    for (int t = 0; t < 4; ++t) {
      const bf16_t* bp = bt + (size_t)(t * 16 + bn) * BROW + bk;
      bb0[t].q[0] = *(const uint4*)(bp + 0);
      bb0[t].q[1] = *(const uint4*)(bp + 8);
      bb1[t].q[0] = *(const uint4*)(bp + 32);
      bb1[t].q[1] = *(const uint4*)(bp + 40);
    }
#pragma unroll
    for (int t = 0; t < 4; ++t)
      c[t] = __builtin_amdgcn_wmma_f32_16x16x32_bf16(
          false, a0.v, false, bb0[t].v, (short)0, c[t], false, false);
#pragma unroll
    for (int t = 0; t < 4; ++t)
      c[t] = __builtin_amdgcn_wmma_f32_16x16x32_bf16(
          false, a1.v, false, bb1[t].v, (short)0, c[t], false, false);

    if (w == 0 && ch + 1 < nch) __builtin_amdgcn_s_wait_tensorcnt((short)0);
    __syncthreads();  // next TDM write to buf^1 only starts after this fence
  }

#pragma unroll
  for (int t = 0; t < 4; ++t) {
    if constexpr (TRANS) {
      // out[b][h][s]: lane's 8 C rows are contiguous s -> one b128 store.
      PackBF8 o;
#pragma unroll
      for (int i = 0; i < 8; ++i) o.v[i] = (bf16_t)c[t][i];
      const int hg = n0 + t * 16 + cn;
      const int bb = m0 / batchRows;
      const int sl = (m0 % batchRows) + cm;
      bf16_t* op = (bf16_t*)Out + (size_t)bb * oBatch + (size_t)hg * ldo + sl;
      *(uint4*)op = o.q;
    } else {
      // LDS micro-transpose: column-layout C -> row-major global stores.
      float* tl = &tileLds[w][0];
#pragma unroll
      for (int i = 0; i < 8; ++i) tl[(cm + i) * 20 + cn] = c[t][i];
      __builtin_amdgcn_wave_barrier();  // keep ds_store before ds_load
      float4 f0 = *(const float4*)&tl[rr * 20 + rh];
      float4 f1 = *(const float4*)&tl[rr * 20 + rh + 4];
      __builtin_amdgcn_wave_barrier();  // keep loads before next tile's stores
      const size_t row = (size_t)(m0 + rr);
      const size_t col = (size_t)(n0 + t * 16 + rh);
      if constexpr (OUT_BF16) {
        PackBF8 o;
        o.v[0] = (bf16_t)f0.x; o.v[1] = (bf16_t)f0.y;
        o.v[2] = (bf16_t)f0.z; o.v[3] = (bf16_t)f0.w;
        o.v[4] = (bf16_t)f1.x; o.v[5] = (bf16_t)f1.y;
        o.v[6] = (bf16_t)f1.z; o.v[7] = (bf16_t)f1.w;
        bf16_t* op = (bf16_t*)Out + z * oBatch + row * ldo + col;
        *(uint4*)op = o.q;
      } else {
        float* op = (float*)Out + z * oBatch + row * ldo + col;
        *(float4*)(op + 0) = f0;
        *(float4*)(op + 4) = f1;
      }
    }
  }
}

// ---------------------------------------------------------------------------
// Fused scores + softmax.  One WG = 16 query rows x full S=2048.
// 8 waves x (2 passes x 8 tiles) of v_wmma_f32_16x16x32_bf16 over H.
// Score stripe lives in LDS (16 x 2049 f32 = 128KB; pad -> conflict-free
// column reads).  Outputs: attn^T f32 (b,s,k) -> d_out, attn bf16 -> ws.
// ---------------------------------------------------------------------------
#define SROW 2049
__global__ void attn_softmax(const bf16_t* __restrict__ qproj,
                             const bf16_t* __restrict__ kproj,
                             bf16_t* __restrict__ attnbf,
                             float* __restrict__ attnT) {
  extern __shared__ float smem[];      // 16 * SROW floats
  __shared__ float red[16][16];
  __shared__ float rowstat[16];

  const int tid = threadIdx.x;
  const int l   = tid & 31;
  const int w   = tid >> 5;
  const int kq0 = blockIdx.x * 16;
  const int b   = blockIdx.y;

  const bf16_t* qb = qproj + ((size_t)b * Kn + kq0) * Hn;
  const bf16_t* kb = kproj + (size_t)b * Sn * Hn;

  const int am  = l & 15;
  const int ah8 = (l >> 4) * 8;
  const int bn  = l & 15;
  const int bk  = (l >> 4) * 16;
  const int cn  = l & 15;
  const int cm  = (l >> 4) * 8;
  const float scale = 0.03125f;  // rsqrt(1024)

  for (int pass = 0; pass < 2; ++pass) {
    const int sbase = w * 256 + pass * 128;
    v8f c[8] = {};
    const bf16_t* arow = qb + (size_t)am * Hn;
    for (int kk = 0; kk < Hn; kk += 32) {
      FragBF a;
      a.q[0] = *(const uint4*)(arow + kk + 2 * ah8);
      a.q[1] = *(const uint4*)(arow + kk + 8 + 2 * ah8);
      FragBF bb[8];
#pragma unroll
      for (int t = 0; t < 8; ++t) {  // all loads first -> staggered waits
        const bf16_t* lp = kb + (size_t)(sbase + t * 16 + bn) * Hn + (kk + bk);
        bb[t].q[0] = ((const uint4*)lp)[0];
        bb[t].q[1] = ((const uint4*)lp)[1];
      }
#pragma unroll
      for (int t = 0; t < 8; ++t)
        c[t] = __builtin_amdgcn_wmma_f32_16x16x32_bf16(
            false, a.v, false, bb[t].v, (short)0, c[t], false, false);
    }
#pragma unroll
    for (int t = 0; t < 8; ++t)
#pragma unroll
      for (int i = 0; i < 8; ++i)
        smem[(cm + i) * SROW + sbase + t * 16 + cn] = c[t][i] * scale;
  }
  __syncthreads();

  // Two-level row max over 2048 cols: 16 threads/row, 128 elems each.
  const int r   = tid >> 4;
  const int seg = tid & 15;
  float* rowp = &smem[r * SROW + seg * 128];
  float mx = -3.402823e38f;
  for (int j = 0; j < 128; ++j) mx = fmaxf(mx, rowp[j]);
  red[r][seg] = mx;
  __syncthreads();
  if (tid < 16) {
    float m2 = red[tid][0];
    for (int j = 1; j < 16; ++j) m2 = fmaxf(m2, red[tid][j]);
    rowstat[tid] = m2;
  }
  __syncthreads();

  const float rm = rowstat[r];
  float sum = 0.f;
  for (int j = 0; j < 128; ++j) {
    float e = __expf(rowp[j] - rm);
    rowp[j] = e;           // overwrite score with un-normalized exp
    sum += e;
  }
  red[r][seg] = sum;
  __syncthreads();
  if (tid < 16) {
    float s2 = 0.f;
    for (int j = 0; j < 16; ++j) s2 += red[tid][j];
    rowstat[tid] = 1.0f / s2;  // store reciprocal
  }
  __syncthreads();

  // attn (bf16, (b,k,s)): contiguous 8-wide b128 stores per thread.
  const float inv = rowstat[r];
  bf16_t* ab = attnbf + ((size_t)b * Kn + kq0 + r) * Sn + seg * 128;
  for (int j0 = 0; j0 < 128; j0 += 8) {
    PackBF8 o;
#pragma unroll
    for (int i = 0; i < 8; ++i) o.v[i] = (bf16_t)(rowp[j0 + i] * inv);
    *(uint4*)(ab + j0) = o.q;
  }

  // attn^T (f32, (b,s,k)): 16 contiguous floats per s (4 x float4 stores);
  // LDS column reads conflict-free thanks to SROW=2049.
  float* aT = attnT + (size_t)b * Sn * Kn + kq0;
  for (int s = tid; s < Sn; s += 256) {
    float v0[16];
#pragma unroll
    for (int r2 = 0; r2 < 16; ++r2) v0[r2] = smem[r2 * SROW + s] * rowstat[r2];
    float* op = aT + (size_t)s * Kn;
    *(float4*)(op + 0)  = make_float4(v0[0],  v0[1],  v0[2],  v0[3]);
    *(float4*)(op + 4)  = make_float4(v0[4],  v0[5],  v0[6],  v0[7]);
    *(float4*)(op + 8)  = make_float4(v0[8],  v0[9],  v0[10], v0[11]);
    *(float4*)(op + 12) = make_float4(v0[12], v0[13], v0[14], v0[15]);
  }
}

// ---------------------------------------------------------------------------
// Launch.  Workspace layout (bf16 elements), ~134 MB total:
//   [qbf|kbf|vbf : 3*8388608] [Wq|Wk|Wv : 3*1048576]
//   [qproj|kproj : 2*8388608] [vT (B,H,S) : 8388608] [attn (B,K,S) : 16777216]
// d_out: context f32 (B*K*H = 8388608) then attn^T f32 (B*S*K = 16777216).
// ---------------------------------------------------------------------------
extern "C" void kernel_launch(void* const* d_in, const int* in_sizes, int n_in,
                              void* d_out, int out_size, void* d_ws,
                              size_t ws_size, hipStream_t stream) {
  (void)in_sizes; (void)n_in; (void)out_size; (void)ws_size;
  const float* queries = (const float*)d_in[0];
  const float* keys    = (const float*)d_in[1];
  const float* values  = (const float*)d_in[2];
  const float* Wq      = (const float*)d_in[3];
  const float* Wk      = (const float*)d_in[4];
  const float* Wv      = (const float*)d_in[5];
  float* out = (float*)d_out;

  const size_t nIn = (size_t)Bn * Kn * Hn;  // 8388608
  const size_t nW  = (size_t)Hn * Hn;       // 1048576

  bf16_t* ws  = (bf16_t*)d_ws;
  bf16_t* qbf = ws;
  bf16_t* kbf = qbf + nIn;
  bf16_t* vbf = kbf + nIn;
  bf16_t* wqb = vbf + nIn;
  bf16_t* wkb = wqb + nW;
  bf16_t* wvb = wkb + nW;
  bf16_t* qpr = wvb + nW;
  bf16_t* kpr = qpr + nIn;
  bf16_t* vtp = kpr + nIn;  // v projection, transposed (B, H, S)
  bf16_t* abf = vtp + nIn;  // attn weights bf16 (B, K, S)

  dim3 blk(256);

  // 1) cast inputs to bf16
  cast_f32_to_bf16<<<dim3((unsigned)(nIn / 8 / 256)), blk, 0, stream>>>(queries, qbf, (int)(nIn / 8));
  cast_f32_to_bf16<<<dim3((unsigned)(nIn / 8 / 256)), blk, 0, stream>>>(keys,    kbf, (int)(nIn / 8));
  cast_f32_to_bf16<<<dim3((unsigned)(nIn / 8 / 256)), blk, 0, stream>>>(values,  vbf, (int)(nIn / 8));
  cast_f32_to_bf16<<<dim3((unsigned)(nW  / 8 / 256)), blk, 0, stream>>>(Wq, wqb, (int)(nW / 8));
  cast_f32_to_bf16<<<dim3((unsigned)(nW  / 8 / 256)), blk, 0, stream>>>(Wk, wkb, (int)(nW / 8));
  cast_f32_to_bf16<<<dim3((unsigned)(nW  / 8 / 256)), blk, 0, stream>>>(Wv, wvb, (int)(nW / 8));

  // 2) projections: P = X @ W^T  (M = B*2048 = 8192, N = H = 1024)
  gemm_bf16<true, false><<<dim3(16, 64, 1), blk, 0, stream>>>(
      qbf, wqb, qpr, Hn, Hn, Hn, Hn, Kn, 0ULL, 0ULL, 0ULL);
  gemm_bf16<true, false><<<dim3(16, 64, 1), blk, 0, stream>>>(
      kbf, wkb, kpr, Hn, Hn, Hn, Hn, Kn, 0ULL, 0ULL, 0ULL);
  gemm_bf16<true, true><<<dim3(16, 64, 1), blk, 0, stream>>>(
      vbf, wvb, vtp, Hn, Hn, Hn, /*ldo=S*/Sn, /*batchRows=*/Sn,
      0ULL, 0ULL, (unsigned long long)Hn * Sn);

  // 3) fused scores + softmax (dyn LDS: 16 * 2049 * 4 = 131136 B)
  attn_softmax<<<dim3(Kn / 16, Bn), blk, 16 * SROW * sizeof(float), stream>>>(
      qpr, kpr, abf, out + nIn);

  // 4) context = attn @ v  (per-batch: M=2048, N=1024, Kd=2048), f32 -> d_out
  gemm_bf16<false, false><<<dim3(16, 16, Bn), blk, 0, stream>>>(
      abf, vtp, out, Sn, Sn, Sn, Hn, Kn,
      (unsigned long long)Kn * Sn, (unsigned long long)Hn * Sn,
      (unsigned long long)Kn * Hn);
}
